// TchKTNet_10196252360895
// MI455X (gfx1250) — compile-verified, matrix operand order checked by reading, not run on hardware
//
#include <hip/hip_runtime.h>

// ---------------------------------------------------------------------------
// Types for CDNA5 WMMA (wave32)
// ---------------------------------------------------------------------------
typedef __attribute__((ext_vector_type(16))) __bf16 v16bf;
typedef __attribute__((ext_vector_type(8)))  float  v8f;

#define BB 32      // batch
#define SS 200     // seq len
#define T199 199   // S-1
#define MROWS (T199*BB)   // 6368
#define E256 256
#define H256 256

__device__ __forceinline__ float sigmoidf_(float x) {
    return 1.0f / (1.0f + __expf(-x));
}

// Wave-relative LDS byte offset = low 32 bits of the generic (shared-aperture)
// address on amdgcn.
__device__ __forceinline__ unsigned lds_addr32(const void* p) {
    return (unsigned)(unsigned long long)(size_t)p;
}

// Async global -> LDS copy of 16 bytes (CDNA5 GLOBAL_LOAD_ASYNC_TO_LDS_B128,
// tracked by ASYNCcnt). ldsoff is a wave-relative LDS byte offset per lane.
__device__ __forceinline__ void async_copy_b128(unsigned ldsoff,
                                                const void* gptr) {
    asm volatile("global_load_async_to_lds_b128 %0, %1, off"
                 :: "v"(ldsoff), "v"((unsigned long long)(size_t)gptr)
                 : "memory");
}
__device__ __forceinline__ void async_wait0() {
    asm volatile("s_wait_asynccnt 0x0" ::: "memory");
}

// Load a 16x32 bf16 fragment (A layout per ISA 7.12.2; identical addressing
// works for B when the weight is stored row-major [N][K], since lane->n).
// base points at the tile origin (row 0, k 0); ld = row stride in elements.
__device__ __forceinline__ v16bf load_frag(const __bf16* base, int ld) {
    const int lane = threadIdx.x & 31;
    const int ridx = lane & 15;
    const int hh   = lane >> 4;
    union { v16bf v; unsigned u[8]; } f;
    const __bf16* rowp = base + ridx * ld;
#pragma unroll
    for (int r = 0; r < 8; ++r) {
        const int k = ((r < 4) ? (2 * r) : (16 + 2 * (r - 4))) + 8 * hh;
        f.u[r] = *(const unsigned*)(rowp + k);
    }
    return f.v;
}

// ---------------------------------------------------------------------------
// Prep kernels
// ---------------------------------------------------------------------------
__global__ void cvt_f32_bf16_kernel(const float* __restrict__ src,
                                    __bf16* __restrict__ dst, int n) {
    int i = blockIdx.x * blockDim.x + threadIdx.x;
    if (i < n) dst[i] = (__bf16)src[i];
}

__global__ void addvec_kernel(const float* __restrict__ a,
                              const float* __restrict__ b,
                              float* __restrict__ o, int n) {
    int i = blockIdx.x * blockDim.x + threadIdx.x;
    if (i < n) o[i] = a[i] + b[i];
}

// ---------------------------------------------------------------------------
// Embedding: emb_q gather + concept average; writes
//  - emb_qca_current (t-major rows t*32+b, 1024 cols, bf16) for s < 199
//  - h_next cols [0,512)  (rows t*32+b, 768 cols, bf16)     for s >= 1
// ---------------------------------------------------------------------------
__global__ void __launch_bounds__(256)
embed_kernel(const int* __restrict__ q, const int* __restrict__ c,
             const int* __restrict__ r,
             const float* __restrict__ que_emb,
             const float* __restrict__ concept_emb,
             __bf16* __restrict__ qca, __bf16* __restrict__ hnext) {
    const int bs = blockIdx.x;          // 0 .. B*S-1
    const int b = bs / SS;
    const int s = bs % SS;
    const int e = threadIdx.x;          // 0..255

    const int qi = q[b * SS + s];
    const float eq = que_emb[(size_t)qi * E256 + e];

    float sum = 0.f; int cnt = 0;
#pragma unroll
    for (int j = 0; j < 4; ++j) {
        const int ci = c[(b * SS + s) * 4 + j];
        if (ci != -1) { sum += concept_emb[(size_t)ci * E256 + e]; cnt++; }
    }
    const float ec = sum / (float)(cnt > 1 ? cnt : 1);
    const float rf = (float)r[b * SS + s];

    if (s < T199) {
        const size_t row = (size_t)s * BB + b;
        __bf16* p = qca + row * 1024;
        p[e]       = (__bf16)(eq * (1.f - rf));
        p[256 + e] = (__bf16)(ec * (1.f - rf));
        p[512 + e] = (__bf16)(eq * rf);
        p[768 + e] = (__bf16)(ec * rf);
    }
    if (s >= 1) {
        const size_t row = (size_t)(s - 1) * BB + b;
        __bf16* p = hnext + row * 768;
        p[e]       = (__bf16)eq;
        p[256 + e] = (__bf16)ec;
    }
}

// ---------------------------------------------------------------------------
// Generic WMMA GEMM:  C[M,N] = A[M,K](bf16) * W[N,K](bf16)^T + bias, opt ReLU
// Block tile 128x64, 8 waves (4x2) of 32x32 each, K step 32.
// Tiles staged with GLOBAL_LOAD_ASYNC_TO_LDS_B128 (no VGPR round-trip).
// A-tile loads are deliberately unguarded on M: out-of-range rows only feed
// output rows that are masked at the store, and every A source buffer has
// later workspace regions behind it, so the over-read stays in-bounds.
// Output either f32 or bf16 (one of outF/outB non-null).
// ---------------------------------------------------------------------------
__global__ void __launch_bounds__(256)
gemm_bf16_kernel(const __bf16* __restrict__ A, const __bf16* __restrict__ W,
                 const float* __restrict__ bias,
                 float* __restrict__ outF, __bf16* __restrict__ outB,
                 int M, int N, int K, int relu) {
    __shared__ __bf16 As[128 * 32];   // 8KB
    __shared__ __bf16 Ws[64 * 32];    // 4KB

    const int m0 = blockIdx.x * 128;
    const int n0 = blockIdx.y * 64;
    const int tid = threadIdx.x;
    const int wave = tid >> 5;
    const int wm = wave & 3;    // 4 waves along M
    const int wn = wave >> 2;   // 2 waves along N
    const int lane = tid & 31;
    const int hh = lane >> 4;
    const int nl = lane & 15;

    const unsigned ldsA = lds_addr32(As);
    const unsigned ldsW = lds_addr32(Ws);

    v8f acc[2][2] = {};

    for (int k0 = 0; k0 < K; k0 += 32) {
        // A tile (128x32 bf16 = 8KB): 512 x 16B chunks, 2 per thread
#pragma unroll
        for (int i = tid; i < 512; i += 256) {
            const int row = i >> 2;            // 0..127
            const int col = (i & 3) * 8;       // element offset in row
            async_copy_b128(ldsA + (unsigned)((row * 32 + col) * 2),
                            A + (size_t)(m0 + row) * K + k0 + col);
        }
        // W tile (64x32 bf16 = 4KB): 256 x 16B chunks, 1 per thread
        {
            const int row = tid >> 2;
            const int col = (tid & 3) * 8;
            async_copy_b128(ldsW + (unsigned)((row * 32 + col) * 2),
                            W + (size_t)(n0 + row) * K + k0 + col);
        }
        async_wait0();
        __syncthreads();

        v16bf a0 = load_frag(&As[(wm * 32 + 0)  * 32], 32);
        v16bf a1 = load_frag(&As[(wm * 32 + 16) * 32], 32);
        v16bf b0 = load_frag(&Ws[(wn * 32 + 0)  * 32], 32);
        v16bf b1 = load_frag(&Ws[(wn * 32 + 16) * 32], 32);

        acc[0][0] = __builtin_amdgcn_wmma_f32_16x16x32_bf16(false, a0, false, b0, (short)0, acc[0][0], false, false);
        acc[0][1] = __builtin_amdgcn_wmma_f32_16x16x32_bf16(false, a0, false, b1, (short)0, acc[0][1], false, false);
        acc[1][0] = __builtin_amdgcn_wmma_f32_16x16x32_bf16(false, a1, false, b0, (short)0, acc[1][0], false, false);
        acc[1][1] = __builtin_amdgcn_wmma_f32_16x16x32_bf16(false, a1, false, b1, (short)0, acc[1][1], false, false);
        __syncthreads();
    }

    // Epilogue: element (lane, v) of tile (mt, nt) -> (m = v+8*hh, n = nl)
#pragma unroll
    for (int mt = 0; mt < 2; ++mt)
#pragma unroll
    for (int nt = 0; nt < 2; ++nt)
#pragma unroll
    for (int v = 0; v < 8; ++v) {
        const int m = m0 + wm * 32 + mt * 16 + v + 8 * hh;
        const int n = n0 + wn * 32 + nt * 16 + nl;
        if (m < M) {
            float x = acc[mt][nt][v];
            if (bias) x += bias[n];
            if (relu) x = fmaxf(x, 0.f);
            if (outF) outF[(size_t)m * N + n] = x;
            else      outB[(size_t)m * N + n] = (__bf16)x;
        }
    }
}

// ---------------------------------------------------------------------------
// Recurrent LSTM. Single workgroup, 512 threads = 16 waves.
// Wave w owns hidden columns [w*16, w*16+16): its 4 gate tiles (i,f,g,o)
// all live in-wave; cell state stays in registers across all 199 steps.
// gates_x already contains x@Wih^T + b_ih + b_hh.  Next step's gates_x slice
// is prefetched (global_prefetch_b8) while the current step's WMMAs run.
// ---------------------------------------------------------------------------
__global__ void __launch_bounds__(512)
lstm_kernel(const float* __restrict__ gates_x,   // (199*32) x 1024 f32
            const __bf16* __restrict__ whh,      // 1024 x 256 bf16 (row = out)
            __bf16* __restrict__ h_seq,          // (199*32) x 256 bf16
            __bf16* __restrict__ h_next) {       // (199*32) x 768, cols 512..767
    __shared__ __bf16 h_lds[BB * H256];          // 16KB, current h (bf16)

    const int tid = threadIdx.x;
    const int wave = tid >> 5;      // 0..15
    const int lane = tid & 31;
    const int hh = lane >> 4;
    const int nl = lane & 15;
    const int n0 = wave * 16;       // hidden column base for this wave

    // zero h (h0 = 0)
    {
        unsigned* hz = (unsigned*)h_lds;
        for (int i = tid; i < BB * H256 / 2; i += 512) hz[i] = 0u;
    }
    float cst[2][8];                // cell state: (m = mt*16 + v + 8*hh, n = n0+nl)
#pragma unroll
    for (int mt = 0; mt < 2; ++mt)
#pragma unroll
    for (int v = 0; v < 8; ++v) cst[mt][v] = 0.f;
    __syncthreads();

    for (int t = 0; t < T199; ++t) {
        v8f acc[2][4];
        // init accumulators with precomputed x-part (+ both biases)
#pragma unroll
        for (int mt = 0; mt < 2; ++mt)
#pragma unroll
        for (int g = 0; g < 4; ++g) {
            v8f a;
#pragma unroll
            for (int v = 0; v < 8; ++v) {
                const int m = mt * 16 + v + 8 * hh;
                const int n = g * 256 + n0 + nl;
                a[v] = gates_x[((size_t)t * BB + m) * 1024 + n];
            }
            acc[mt][g] = a;
        }

        // prefetch next step's x-gates slice (128KB: 512 thr x 256B)
        if (t + 1 < T199) {
            const float* nxt = gates_x + ((size_t)(t + 1) * BB) * 1024;
            __builtin_prefetch(nxt + (size_t)tid * 64, 0, 0);
        }

        // gates += h @ Whh^T  (K = 256, 8 k-steps of 32)
#pragma unroll
        for (int kt = 0; kt < 8; ++kt) {
            v16bf af[2];
            af[0] = load_frag(&h_lds[(0)  * H256 + kt * 32], H256);
            af[1] = load_frag(&h_lds[(16) * H256 + kt * 32], H256);
            v16bf bf[4];
#pragma unroll
            for (int g = 0; g < 4; ++g)
                bf[g] = load_frag(whh + (size_t)(g * 256 + n0) * H256 + kt * 32, H256);
#pragma unroll
            for (int mt = 0; mt < 2; ++mt)
#pragma unroll
            for (int g = 0; g < 4; ++g)
                acc[mt][g] = __builtin_amdgcn_wmma_f32_16x16x32_bf16(
                    false, af[mt], false, bf[g], (short)0, acc[mt][g], false, false);
        }
        __syncthreads();   // everyone done reading h_lds for this step

        // cell update + write new h
#pragma unroll
        for (int mt = 0; mt < 2; ++mt)
#pragma unroll
        for (int v = 0; v < 8; ++v) {
            const float i_ = acc[mt][0][v];
            const float f_ = acc[mt][1][v];
            const float g_ = acc[mt][2][v];
            const float o_ = acc[mt][3][v];
            float cs = sigmoidf_(f_) * cst[mt][v] + sigmoidf_(i_) * tanhf(g_);
            cst[mt][v] = cs;
            const float hv = sigmoidf_(o_) * tanhf(cs);
            const __bf16 hb = (__bf16)hv;
            const int m = mt * 16 + v + 8 * hh;
            const int n = n0 + nl;
            h_lds[m * H256 + n] = hb;
            const size_t row = (size_t)t * BB + m;
            h_seq[row * H256 + n]        = hb;
            h_next[row * 768 + 512 + n]  = hb;
        }
        __syncthreads();   // h writes visible before next step's reads
    }
}

// ---------------------------------------------------------------------------
// Heads: one wave per (b,t). Gathered output rows only (never the full
// NUM_Q/NUM_C projections). Lane-parallel dots + xor-shuffle reduction.
// ---------------------------------------------------------------------------
__device__ __forceinline__ float wave_dot(const __bf16* __restrict__ h,
                                          const float* __restrict__ w,
                                          int L, int lane) {
    float s = 0.f;
    for (int k = lane; k < L; k += 32) s += (float)h[k] * w[k];
#pragma unroll
    for (int m = 16; m > 0; m >>= 1) s += __shfl_xor(s, m, 32);
    return s;
}

__global__ void __launch_bounds__(256)
heads_kernel(const __bf16* __restrict__ hid_qn, const __bf16* __restrict__ hid_cn,
             const __bf16* __restrict__ hid_qa, const __bf16* __restrict__ hid_ca,
             const float* __restrict__ qn_ow, const float* __restrict__ qn_ob,
             const float* __restrict__ cn_ow, const float* __restrict__ cn_ob,
             const float* __restrict__ qa_ow, const float* __restrict__ qa_ob,
             const float* __restrict__ ca_ow, const float* __restrict__ ca_ob,
             const int* __restrict__ qshft, const int* __restrict__ cshft,
             float* __restrict__ out) {
    const int row = (blockIdx.x * blockDim.x + threadIdx.x) >> 5;  // wave id
    if (row >= MROWS) return;
    const int lane = threadIdx.x & 31;
    const int t = row >> 5;       // rows are t*32 + b
    const int b = row & 31;

    // y_question_next
    const float yqn = sigmoidf_(
        wave_dot(hid_qn + (size_t)row * 768, qn_ow, 768, lane) + qn_ob[0]);

    // y_concept_next (avg over masked gathered concepts)
    float accn = 0.f; int cntn = 0;
#pragma unroll
    for (int j = 0; j < 4; ++j) {
        const int ci = cshft[(b * T199 + t) * 4 + j];   // wave-uniform
        if (ci != -1) {
            const float d = wave_dot(hid_cn + (size_t)row * 768,
                                     cn_ow + (size_t)ci * 768, 768, lane) + cn_ob[ci];
            accn += sigmoidf_(d); cntn++;
        }
    }
    const float ycn = accn / (float)(cntn > 1 ? cntn : 1);

    // y_question_all (single gathered row of the 12000-wide projection)
    const int qi = qshft[b * T199 + t];
    const float yqa = sigmoidf_(
        wave_dot(hid_qa + (size_t)row * 256, qa_ow + (size_t)qi * 256, 256, lane)
        + qa_ob[qi]);

    // y_concept_all
    float acca = 0.f; int cnta = 0;
#pragma unroll
    for (int j = 0; j < 4; ++j) {
        const int ci = cshft[(b * T199 + t) * 4 + j];
        if (ci != -1) {
            const float d = wave_dot(hid_ca + (size_t)row * 256,
                                     ca_ow + (size_t)ci * 256, 256, lane) + ca_ob[ci];
            acca += sigmoidf_(d); cnta++;
        }
    }
    const float yca = acca / (float)(cnta > 1 ? cnta : 1);

    if (lane == 0) {
        const int oidx = b * T199 + t;
        out[oidx]             = yqn;
        out[MROWS + oidx]     = ycn;
        out[2 * MROWS + oidx] = yqa;
        out[3 * MROWS + oidx] = yca;
    }
}

// ---------------------------------------------------------------------------
// Host launcher
// ---------------------------------------------------------------------------
extern "C" void kernel_launch(void* const* d_in, const int* in_sizes, int n_in,
                              void* d_out, int out_size, void* d_ws, size_t ws_size,
                              hipStream_t stream) {
    (void)in_sizes; (void)n_in; (void)out_size; (void)ws_size;

    const int*   q           = (const int*)d_in[0];
    const int*   c           = (const int*)d_in[1];
    const int*   r           = (const int*)d_in[2];
    const int*   qshft       = (const int*)d_in[3];
    const int*   cshft       = (const int*)d_in[4];
    const float* que_emb     = (const float*)d_in[5];
    const float* concept_emb = (const float*)d_in[6];
    const float* w_ih        = (const float*)d_in[7];
    const float* w_hh        = (const float*)d_in[8];
    const float* b_ih        = (const float*)d_in[9];
    const float* b_hh        = (const float*)d_in[10];
    const float* qn_lw = (const float*)d_in[11]; const float* qn_lb = (const float*)d_in[12];
    const float* qn_ow = (const float*)d_in[13]; const float* qn_ob = (const float*)d_in[14];
    const float* cn_lw = (const float*)d_in[15]; const float* cn_lb = (const float*)d_in[16];
    const float* cn_ow = (const float*)d_in[17]; const float* cn_ob = (const float*)d_in[18];
    const float* qa_lw = (const float*)d_in[19]; const float* qa_lb = (const float*)d_in[20];
    const float* qa_ow = (const float*)d_in[21]; const float* qa_ob = (const float*)d_in[22];
    const float* ca_lw = (const float*)d_in[23]; const float* ca_lb = (const float*)d_in[24];
    const float* ca_ow = (const float*)d_in[25]; const float* ca_ob = (const float*)d_in[26];
    float* out = (float*)d_out;

    // workspace carve-up (256B aligned)
    char* ws = (char*)d_ws;
    size_t off = 0;
    auto carve = [&](size_t bytes) -> char* {
        char* p = ws + off;
        off += (bytes + 255) & ~(size_t)255;
        return p;
    };
    __bf16* w_ih_b  = (__bf16*)carve((size_t)1024 * 1024 * 2);
    __bf16* w_hh_b  = (__bf16*)carve((size_t)1024 * 256 * 2);
    __bf16* qn_lw_b = (__bf16*)carve((size_t)768 * 768 * 2);
    __bf16* cn_lw_b = (__bf16*)carve((size_t)768 * 768 * 2);
    __bf16* qa_lw_b = (__bf16*)carve((size_t)256 * 256 * 2);
    __bf16* ca_lw_b = (__bf16*)carve((size_t)256 * 256 * 2);
    float*  b_comb  = (float*) carve((size_t)1024 * 4);
    __bf16* qca     = (__bf16*)carve((size_t)MROWS * 1024 * 2);
    __bf16* hnext   = (__bf16*)carve((size_t)MROWS * 768 * 2);
    float*  gates_x = (float*) carve((size_t)MROWS * 1024 * 4);
    __bf16* h_seq   = (__bf16*)carve((size_t)MROWS * 256 * 2);
    __bf16* hid_qn  = (__bf16*)carve((size_t)MROWS * 768 * 2);
    __bf16* hid_cn  = (__bf16*)carve((size_t)MROWS * 768 * 2);
    __bf16* hid_qa  = (__bf16*)carve((size_t)MROWS * 256 * 2);
    __bf16* hid_ca  = (__bf16*)carve((size_t)MROWS * 256 * 2);

    // 1) weight conversions to bf16
    auto cvt = [&](const float* s, __bf16* d, int n) {
        cvt_f32_bf16_kernel<<<(n + 255) / 256, 256, 0, stream>>>(s, d, n);
    };
    cvt(w_ih,  w_ih_b,  1024 * 1024);
    cvt(w_hh,  w_hh_b,  1024 * 256);
    cvt(qn_lw, qn_lw_b, 768 * 768);
    cvt(cn_lw, cn_lw_b, 768 * 768);
    cvt(qa_lw, qa_lw_b, 256 * 256);
    cvt(ca_lw, ca_lw_b, 256 * 256);
    addvec_kernel<<<4, 256, 0, stream>>>(b_ih, b_hh, b_comb, 1024);

    // 2) embeddings -> emb_qca_current (t-major) and h_next[:, 0:512]
    embed_kernel<<<BB * SS, 256, 0, stream>>>(q, c, r, que_emb, concept_emb,
                                              qca, hnext);

    // 3) time-parallel input projection: gates_x = qca @ w_ih^T + (b_ih + b_hh)
    {
        dim3 grid((MROWS + 127) / 128, 1024 / 64);
        gemm_bf16_kernel<<<grid, 256, 0, stream>>>(
            qca, w_ih_b, b_comb, gates_x, nullptr, MROWS, 1024, 1024, 0);
    }

    // 4) sequential LSTM recurrence (single workgroup, WMMA per step)
    lstm_kernel<<<1, 512, 0, stream>>>(gates_x, w_hh_b, h_seq, hnext);

    // 5) MLP hidden layers (ReLU), bf16 outputs
    {
        dim3 grid((MROWS + 127) / 128, 768 / 64);
        gemm_bf16_kernel<<<grid, 256, 0, stream>>>(
            hnext, qn_lw_b, qn_lb, nullptr, hid_qn, MROWS, 768, 768, 1);
        gemm_bf16_kernel<<<grid, 256, 0, stream>>>(
            hnext, cn_lw_b, cn_lb, nullptr, hid_cn, MROWS, 768, 768, 1);
    }
    {
        dim3 grid((MROWS + 127) / 128, 256 / 64);
        gemm_bf16_kernel<<<grid, 256, 0, stream>>>(
            h_seq, qa_lw_b, qa_lb, nullptr, hid_qa, MROWS, 256, 256, 1);
        gemm_bf16_kernel<<<grid, 256, 0, stream>>>(
            h_seq, ca_lw_b, ca_lb, nullptr, hid_ca, MROWS, 256, 256, 1);
    }

    // 6) gathered output heads (sigmoid + masked averaging), one wave per (b,t)
    heads_kernel<<<(MROWS * 32 + 255) / 256, 256, 0, stream>>>(
        hid_qn, hid_cn, hid_qa, hid_ca,
        qn_ow, qn_ob, cn_ow, cn_ob, qa_ow, qa_ob, ca_ow, ca_ob,
        qshft, cshft, out);
}
